// ModernCausalSelfAttention_70085276336444
// MI455X (gfx1250) — matrix operand order repaired and checked by reading, not compiled, not run on hardware
//
#include <hip/hip_runtime.h>

#define B_ 2
#define S_ 2048
#define H_ 16
#define G_ 4
#define D_ 128
#define DM 2048
#define BS (B_*S_)

typedef __attribute__((ext_vector_type(16))) __bf16 v16bf;
typedef __attribute__((ext_vector_type(8)))  __bf16 v8bf;
typedef __attribute__((ext_vector_type(8)))  float  v8f;

#define WMMA_BF16(A,Bm,Cm) \
  __builtin_amdgcn_wmma_f32_16x16x32_bf16(false,(A),false,(Bm),(short)0,(Cm),false,false)

static __device__ __forceinline__ v8f zero8() {
  v8f z = {0.f,0.f,0.f,0.f,0.f,0.f,0.f,0.f};
  return z;
}

// A-matrix 16x32 bf16 fragment.  Layout: lanes 0-15 hold K = k0..k0+15;
// lanes 16-31 hold K = k0+16..k0+31 (contiguous, matching frag_b).
static __device__ __forceinline__ v16bf frag_a(const __bf16* base, int stride,
                                               int r0, int k0) {
  const int lane = threadIdx.x & 31;
  const int half = lane >> 4;
  const __bf16* p = base + (size_t)(r0 + (lane & 15)) * stride + k0 + half * 16;
  v8bf lo = *(const v8bf*)p;        // K = k0 + half*16 + 0..7
  v8bf hi = *(const v8bf*)(p + 8);  // K = k0 + half*16 + 8..15
  return __builtin_shufflevector(lo, hi, 0,1,2,3,4,5,6,7,8,9,10,11,12,13,14,15);
}

// B-matrix 32x16 bf16 fragment, sourced from row-major Bt[N,K] (i.e. column n
// of B is row n of Bt).  Layout used here: lane holds column n = lane%16; lanes 0-15
// hold K = k0..k0+15, lanes 16-31 hold K = k0+16..k0+31 (contiguous 32B).
static __device__ __forceinline__ v16bf frag_b(const __bf16* base, int stride,
                                               int n0, int k0) {
  const int lane = threadIdx.x & 31;
  const int half = lane >> 4;
  const __bf16* p = base + (size_t)(n0 + (lane & 15)) * stride + k0 + half * 16;
  v8bf lo = *(const v8bf*)p;
  v8bf hi = *(const v8bf*)(p + 8);
  return __builtin_shufflevector(lo, hi, 0,1,2,3,4,5,6,7,8,9,10,11,12,13,14,15);
}

static __device__ __forceinline__ float rmax16(float v) {
  v = fmaxf(v, __shfl_xor(v, 1));
  v = fmaxf(v, __shfl_xor(v, 2));
  v = fmaxf(v, __shfl_xor(v, 4));
  v = fmaxf(v, __shfl_xor(v, 8));
  return v;
}
static __device__ __forceinline__ float rsum16(float v) {
  v += __shfl_xor(v, 1);
  v += __shfl_xor(v, 2);
  v += __shfl_xor(v, 4);
  v += __shfl_xor(v, 8);
  return v;
}

// ---------------- conversion / layout kernels ----------------

__global__ void cvt_f32_bf16(const float* __restrict__ in, __bf16* __restrict__ out, int n) {
  int i = blockIdx.x * blockDim.x + threadIdx.x;
  if (i < n) out[i] = (__bf16)in[i];
}

// in: f32 [R,C] row-major  ->  out: bf16 [C,R] row-major (Bt layout for GEMM)
__global__ void transpose_f32_bf16(const float* __restrict__ in, __bf16* __restrict__ out,
                                   int R, int C) {
  int i = blockIdx.x * blockDim.x + threadIdx.x;
  if (i >= R * C) return;
  int r = i / C, c = i - r * C;
  out[(size_t)c * R + r] = (__bf16)in[i];
}

// src: f32 [B*S, NH*128] (GEMM output) -> dst: bf16 [B, NH, S, 128] with
// interleaved RoPE applied:  out[2i] = x0*cos - x1*sin ; out[2i+1] = x1*cos + x0*sin
__global__ void rope_pack(const float* __restrict__ src, __bf16* __restrict__ dst, int NH) {
  int i = blockIdx.x * blockDim.x + threadIdx.x;
  int total = B_ * S_ * NH * (D_ / 2);
  if (i >= total) return;
  int p = i & 63;
  int h = (i >> 6) % NH;
  int s = (i / (64 * NH)) % S_;
  int b = i / (64 * NH * S_);
  // inv_freq = 10000^(-2p/128)
  float inv = __expf(-(float)(2 * p) * (0.0078125f * 9.210340371976184f)); // ln(10000)/128
  float ang = (float)s * inv;
  float sn, cs;
  sincosf(ang, &sn, &cs);
  const float* row = src + (size_t)(b * S_ + s) * (NH * D_) + h * D_;
  float x0 = row[2 * p], x1 = row[2 * p + 1];
  __bf16* o = dst + ((size_t)(b * NH + h) * S_ + s) * D_;
  o[2 * p]     = (__bf16)(x0 * cs - x1 * sn);
  o[2 * p + 1] = (__bf16)(x1 * cs + x0 * sn);
}

// vf: f32 [B*S, G*128] -> vt: bf16 [B, G, 128, S]  (V transposed per head)
__global__ void pack_vt(const float* __restrict__ vf, __bf16* __restrict__ vt) {
  int i = blockIdx.x * blockDim.x + threadIdx.x;
  int total = B_ * S_ * G_ * D_;
  if (i >= total) return;
  int d = i & 127;
  int g = (i >> 7) & 3;
  int s = (i >> 9) % S_;
  int b = i / (512 * S_);
  vt[((size_t)(b * G_ + g) * D_ + d) * S_ + s] = (__bf16)vf[i];
}

// ---------------- WMMA GEMM:  C[M,N] f32 = A[M,K] bf16 @ Bt[N,K] bf16 ----------------
// 256 threads = 8 waves; block computes 256(M) x 64(N); each wave one 32x64 tile
// (2 A-frags x 4 B-frags = 8 WMMAs per 32-wide K step).
__global__ __launch_bounds__(256) void gemm_bf16_wmma(const __bf16* __restrict__ A,
                                                      const __bf16* __restrict__ Bt,
                                                      float* __restrict__ C,
                                                      int M, int N, int K) {
  const int wave = threadIdx.x >> 5, lane = threadIdx.x & 31;
  const int half = lane >> 4, col = lane & 15;
  const int m0 = (blockIdx.y * 8 + wave) * 32;
  const int n0 = blockIdx.x * 64;
  if (m0 >= M) return;
  v8f acc[2][4];
#pragma unroll
  for (int i = 0; i < 2; ++i)
#pragma unroll
    for (int j = 0; j < 4; ++j) acc[i][j] = zero8();

  const __bf16* arow0 = A + (size_t)(m0 + (lane & 15)) * K;
  const __bf16* arow1 = A + (size_t)(m0 + 16 + (lane & 15)) * K;
  const __bf16* brow  = Bt + (size_t)(n0 + (lane & 15)) * K;

  for (int k0 = 0; k0 < K; k0 += 32) {
    // pull the next K-tile of this lane's rows toward the WGP (global_prefetch_b8;
    // no LOADcnt traffic per ISA 10.5)
    if (k0 + 64 <= K) {
      __builtin_prefetch(arow0 + k0 + 32, 0, 3);
      __builtin_prefetch(arow1 + k0 + 32, 0, 3);
      __builtin_prefetch(brow + k0 + 32, 0, 3);
      __builtin_prefetch(brow + 16 * (size_t)K + k0 + 32, 0, 3);
      __builtin_prefetch(brow + 32 * (size_t)K + k0 + 32, 0, 3);
      __builtin_prefetch(brow + 48 * (size_t)K + k0 + 32, 0, 3);
    }
    v16bf a0 = frag_a(A, K, m0, k0);
    v16bf a1 = frag_a(A, K, m0 + 16, k0);
#pragma unroll
    for (int j = 0; j < 4; ++j) {
      v16bf b = frag_b(Bt, K, n0 + j * 16, k0);
      acc[0][j] = WMMA_BF16(a0, b, acc[0][j]);
      acc[1][j] = WMMA_BF16(a1, b, acc[1][j]);
    }
  }
  // C layout: VGPR r -> row m0 + i*16 + half*8 + r, col n0 + j*16 + lane%16
#pragma unroll
  for (int i = 0; i < 2; ++i)
#pragma unroll
    for (int r = 0; r < 8; ++r) {
      size_t idx = (size_t)(m0 + i * 16 + half * 8 + r) * N + n0 + col;
      C[idx]      = acc[i][0][r];
      C[idx + 16] = acc[i][1][r];
      C[idx + 32] = acc[i][2][r];
      C[idx + 48] = acc[i][3][r];
    }
}

// ---------------- flash attention ----------------
// grid (S/64, H, B), 128 threads = 4 waves; wave owns 16 query rows.
// qb [B,H,S,128], kb [B,G,S,128], vt [B,G,128,S]  -> yb bf16 [B*S, 2048]
__global__ __launch_bounds__(128) void attn_kernel(const __bf16* __restrict__ qb,
                                                   const __bf16* __restrict__ kb,
                                                   const __bf16* __restrict__ vt,
                                                   __bf16* __restrict__ yb) {
  __shared__ __align__(16) __bf16 plds[4][16 * 32];
  const int wave = threadIdx.x >> 5, lane = threadIdx.x & 31;
  const int half = lane >> 4, col = lane & 15;
  const int b = blockIdx.z, h = blockIdx.y, g = h >> 2;
  const int q0 = blockIdx.x * 64 + wave * 16;
  const float scale = 0.08838834764831845f; // 1/sqrt(128)

  const __bf16* qbase = qb + (size_t)(b * H_ + h) * S_ * D_;
  const __bf16* kbase = kb + (size_t)(b * G_ + g) * S_ * D_;
  const __bf16* vbase = vt + (size_t)(b * G_ + g) * D_ * S_;
  __bf16* pl = &plds[wave][0];

  // Q fragments over d = 0..127 (K dimension of QK^T), loaded once
  v16bf qa0 = frag_a(qbase, D_, q0, 0);
  v16bf qa1 = frag_a(qbase, D_, q0, 32);
  v16bf qa2 = frag_a(qbase, D_, q0, 64);
  v16bf qa3 = frag_a(qbase, D_, q0, 96);

  v8f o[8];
#pragma unroll
  for (int j = 0; j < 8; ++j) o[j] = zero8();
  float m8[8], l8[8];
#pragma unroll
  for (int r = 0; r < 8; ++r) { m8[r] = -1e30f; l8[r] = 0.f; }

  const int kend = q0 + 16; // causal: keys < q0+16
  for (int kv0 = 0; kv0 < kend; kv0 += 32) {
    // scores: two 16x16 tiles over this 32-key chunk, K-chained over d=128
    v8f s0 = zero8(), s1 = zero8();
    {
      v16bf k00 = frag_b(kbase, D_, kv0,      0);
      v16bf k10 = frag_b(kbase, D_, kv0 + 16, 0);
      s0 = WMMA_BF16(qa0, k00, s0); s1 = WMMA_BF16(qa0, k10, s1);
      v16bf k01 = frag_b(kbase, D_, kv0,      32);
      v16bf k11 = frag_b(kbase, D_, kv0 + 16, 32);
      s0 = WMMA_BF16(qa1, k01, s0); s1 = WMMA_BF16(qa1, k11, s1);
      v16bf k02 = frag_b(kbase, D_, kv0,      64);
      v16bf k12 = frag_b(kbase, D_, kv0 + 16, 64);
      s0 = WMMA_BF16(qa2, k02, s0); s1 = WMMA_BF16(qa2, k12, s1);
      v16bf k03 = frag_b(kbase, D_, kv0,      96);
      v16bf k13 = frag_b(kbase, D_, kv0 + 16, 96);
      s0 = WMMA_BF16(qa3, k03, s0); s1 = WMMA_BF16(qa3, k13, s1);
    }
    // online softmax (per row r: row = q0 + half*8 + r; this lane's col = key col)
    float alpha[8];
#pragma unroll
    for (int r = 0; r < 8; ++r) {
      int row = q0 + half * 8 + r;
      float a0 = (kv0 + col      > row) ? -1e30f : s0[r] * scale;
      float a1 = (kv0 + 16 + col > row) ? -1e30f : s1[r] * scale;
      float mx = rmax16(fmaxf(a0, a1));
      float nm = fmaxf(m8[r], mx);
      float al = __expf(m8[r] - nm);
      float p0 = __expf(a0 - nm);
      float p1 = __expf(a1 - nm);
      float rs = rsum16(p0 + p1);
      l8[r] = l8[r] * al + rs;
      m8[r] = nm;
      alpha[r] = al;
      // stash P (bf16) in LDS to re-layout C-frag -> A-frag (per-wave, DS in-order)
      pl[(half * 8 + r) * 32 + col]      = (__bf16)p0;
      pl[(half * 8 + r) * 32 + 16 + col] = (__bf16)p1;
    }
#pragma unroll
    for (int j = 0; j < 8; ++j)
#pragma unroll
      for (int r = 0; r < 8; ++r) o[j][r] *= alpha[r];
    // PV: A = P[16x32] from LDS, B = V[32x16] per d-chunk from Vt rows
    v16bf pf = frag_a(pl, 32, 0, 0);
#pragma unroll
    for (int j = 0; j < 8; ++j) {
      v16bf vfg = frag_b(vbase, S_, j * 16, kv0);
      o[j] = WMMA_BF16(pf, vfg, o[j]);
    }
  }

  float invl[8];
#pragma unroll
  for (int r = 0; r < 8; ++r) invl[r] = 1.f / l8[r];
#pragma unroll
  for (int j = 0; j < 8; ++j)
#pragma unroll
    for (int r = 0; r < 8; ++r) {
      size_t idx = (size_t)(b * S_ + q0 + half * 8 + r) * DM + h * D_ + j * 16 + col;
      yb[idx] = (__bf16)(o[j][r] * invl[r]);
    }
}

// ---------------- host-side orchestration ----------------

extern "C" void kernel_launch(void* const* d_in, const int* in_sizes, int n_in,
                              void* d_out, int out_size, void* d_ws, size_t ws_size,
                              hipStream_t stream) {
  const float* x  = (const float*)d_in[0]; // [B,S,2048]
  const float* wq = (const float*)d_in[1]; // [2048,2048]
  const float* wk = (const float*)d_in[2]; // [2048,512]
  const float* wv = (const float*)d_in[3]; // [2048,512]
  const float* wo = (const float*)d_in[4]; // [2048,2048]

  char* w = (char*)d_ws;
  auto alloc = [&](size_t bytes) -> void* {
    void* p = (void*)w;
    w += (bytes + 255) & ~(size_t)255;
    return p;
  };

  __bf16* xb  = (__bf16*)alloc((size_t)BS * DM * 2);          // x in bf16
  __bf16* wqt = (__bf16*)alloc((size_t)DM * DM * 2);          // wq^T
  __bf16* wkt = (__bf16*)alloc((size_t)512 * DM * 2);         // wk^T
  __bf16* wvt = (__bf16*)alloc((size_t)512 * DM * 2);         // wv^T
  __bf16* wot = (__bf16*)alloc((size_t)DM * DM * 2);          // wo^T
  float*  qf  = (float*)alloc((size_t)BS * DM * 4);           // x@wq  f32
  float*  kf  = (float*)alloc((size_t)BS * 512 * 4);          // x@wk  f32
  float*  vf  = (float*)alloc((size_t)BS * 512 * 4);          // x@wv  f32
  __bf16* qb  = (__bf16*)alloc((size_t)BS * DM * 2);          // roped Q [B,H,S,D]
  __bf16* kb  = (__bf16*)alloc((size_t)BS * 512 * 2);         // roped K [B,G,S,D]
  __bf16* vt  = (__bf16*)alloc((size_t)BS * 512 * 2);         // V^T   [B,G,D,S]
  __bf16* yb  = (__bf16*)alloc((size_t)BS * DM * 2);          // attn out bf16

  const int T = 256;
  // 1) convert x
  {
    int n = BS * DM;
    cvt_f32_bf16<<<(n + T - 1) / T, T, 0, stream>>>(x, xb, n);
  }
  // 2) transpose-convert weights into Bt[N,K] layout
  transpose_f32_bf16<<<(DM * DM + T - 1) / T, T, 0, stream>>>(wq, wqt, DM, DM);
  transpose_f32_bf16<<<(DM * 512 + T - 1) / T, T, 0, stream>>>(wk, wkt, DM, 512);
  transpose_f32_bf16<<<(DM * 512 + T - 1) / T, T, 0, stream>>>(wv, wvt, DM, 512);
  transpose_f32_bf16<<<(DM * DM + T - 1) / T, T, 0, stream>>>(wo, wot, DM, DM);
  // 3) QKV projections (block tile 256x64)
  gemm_bf16_wmma<<<dim3(DM / 64, BS / 256), 256, 0, stream>>>(xb, wqt, qf, BS, DM, DM);
  gemm_bf16_wmma<<<dim3(512 / 64, BS / 256), 256, 0, stream>>>(xb, wkt, kf, BS, 512, DM);
  gemm_bf16_wmma<<<dim3(512 / 64, BS / 256), 256, 0, stream>>>(xb, wvt, vf, BS, 512, DM);
  // 4) RoPE + layout
  {
    int nq = B_ * S_ * H_ * (D_ / 2);
    rope_pack<<<(nq + T - 1) / T, T, 0, stream>>>(qf, qb, H_);
    int nk = B_ * S_ * G_ * (D_ / 2);
    rope_pack<<<(nk + T - 1) / T, T, 0, stream>>>(kf, kb, G_);
    int nv = B_ * S_ * G_ * D_;
    pack_vt<<<(nv + T - 1) / T, T, 0, stream>>>(vf, vt);
  }
  // 5) causal flash attention
  attn_kernel<<<dim3(S_ / 64, H_, B_), 128, 0, stream>>>(qb, kb, vt, yb);
  // 6) output projection -> d_out (f32)
  gemm_bf16_wmma<<<dim3(DM / 64, BS / 256), 256, 0, stream>>>(yb, wot, (float*)d_out,
                                                              BS, DM, DM);
}
